// INT4MultiHeadLatentAttention_84516366451355
// MI455X (gfx1250) — matrix operand order repaired and checked by reading, not compiled, not run on hardware
//
#include <hip/hip_runtime.h>

// ---------------------------------------------------------------------------
// INT4 Multi-Head Latent Attention for MI455X (gfx1250), bf16 WMMA pipeline.
// B=2, N=2048, D=2048, H=16, HD=128, R=512.
//
// - All WMMA B-operands kept in "fragment order": 16 bf16 contiguous per lane
//   -> every operand load is a 32B contiguous vector load.
// - All LDS tile staging uses GLOBAL_LOAD_ASYNC_TO_LDS_B128 (ASYNCcnt) when
//   the toolchain exposes it; falls back to 16B register copies otherwise.
// - V projection is written transposed (V^T: dim-major) so V fragments are
//   16B-contiguous in global memory as well.
// ---------------------------------------------------------------------------

#define B_ 2
#define N_ 2048
#define D_ 2048
#define H_ 16
#define HD_ 128
#define R_ 512
#define M_ (B_ * N_)   // 4096 rows in the flattened (B*N, D) view

typedef __attribute__((ext_vector_type(16))) __bf16 bf16x16;
typedef __attribute__((ext_vector_type(8)))  float  f32x8;
typedef __attribute__((ext_vector_type(4)))  int    v4i;

#if defined(__has_builtin)
#if __has_builtin(__builtin_amdgcn_global_load_async_to_lds_b128) && \
    __has_builtin(__builtin_amdgcn_s_wait_asynccnt)
#define HAS_ASYNC_LDS 1
#endif
#endif
#ifndef HAS_ASYNC_LDS
#define HAS_ASYNC_LDS 0
#endif

// ------------------------------ helpers ------------------------------------

static __device__ __forceinline__ __bf16 f2bf(float f) {
  unsigned int u = __builtin_bit_cast(unsigned int, f);
  unsigned int r = u + 0x7fffu + ((u >> 16) & 1u);   // round-to-nearest-even
  unsigned short s = (unsigned short)(r >> 16);
  return __builtin_bit_cast(__bf16, s);
}

// 16-byte tile-staging copy: async DMA to LDS when available.
static __device__ __forceinline__ void async_copy16(void* lds, const void* g) {
#if HAS_ASYNC_LDS
  __builtin_amdgcn_global_load_async_to_lds_b128(
      (__attribute__((address_space(1))) v4i*)g,
      (__attribute__((address_space(3))) v4i*)lds, 0, 0);
#else
  *(int4*)lds = *(const int4*)g;
#endif
}

static __device__ __forceinline__ void async_wait() {
#if HAS_ASYNC_LDS
  __builtin_amdgcn_s_wait_asynccnt(0);
#endif
}

static __device__ __forceinline__ f32x8 wmma_bf16(bf16x16 a, bf16x16 b, f32x8 c) {
  // D = A(16x32 bf16) * B(32x16 bf16) + C(16x16 f32)
  return __builtin_amdgcn_wmma_f32_16x16x32_bf16(false, a, false, b,
                                                 (short)0, c, false, false);
}

// A-fragment (16x32, MxK) from row-major LDS tile, contiguous k order (k = 16*half + e).
// Both 8-element groups are contiguous -> 2x ds_load_b128 per lane.
static __device__ __forceinline__ bf16x16
load_a_frag(const __bf16* sA, int row0, int lda, int k0, int r, int half) {
  bf16x16 f;
  const __bf16* base = sA + (row0 + r) * lda + k0;
#pragma unroll
  for (int e = 0; e < 8; ++e) {
    f[e]     = base[half * 16 + e];
    f[e + 8] = base[half * 16 + 8 + e];
  }
  return f;
}

// B-fragment in fragment-order storage: 16 contiguous bf16 per lane.
static __device__ __forceinline__ bf16x16
load_b_frag_packed(const __bf16* p, int cg, int ks, int nKs, int lane) {
  return *(const bf16x16*)(p + (size_t)(((cg * nKs + ks) * 32) + lane) * 16);
}

// --------------------------- conversion kernels ----------------------------

__global__ __launch_bounds__(256) void f32_to_bf16_kernel(
    const float* __restrict__ in, __bf16* __restrict__ out, int n) {
  int i = blockIdx.x * 256 + threadIdx.x;
  if (i < n) out[i] = f2bf(in[i]);
}

// Row-major f32 weight [K x Nc] -> bf16 WMMA B-fragment order:
//   (krow, n) -> out[((cg*nKs + ks)*32 + lane)*16 + e]
//   ks = krow/32, e = krow%16, lane = ((krow>>4)&1)*16 + (n&15), cg = n/16
__global__ __launch_bounds__(256) void f32_to_bf16_swzB_kernel(
    const float* __restrict__ in, __bf16* __restrict__ out, int K, int Nc) {
  int i = blockIdx.x * 256 + threadIdx.x;
  if (i >= K * Nc) return;
  int krow = i / Nc, n = i - krow * Nc;
  int ks = krow >> 5;
  int e = krow & 15;
  int lane = ((krow >> 4) & 1) * 16 + (n & 15);
  int cg = n >> 4;
  int nKs = K >> 5;
  out[(size_t)(((cg * nKs + ks) * 32) + lane) * 16 + e] = f2bf(in[i]);
}

// ------------------------------- GEMM --------------------------------------
// C[M,Nc] = A[M,K](bf16 rowmajor) @ Bsw[K,Nc](bf16, fragment order) + bias
// Block tile 64x128, K-step 32, 256 threads (8 waves), wave tile 32x32.
// A staged to LDS via async DMA; B fragments loaded straight from global.
// TRANS_OUT: write C^T (bf16) at C[col*M + row]  (used for V^T).

template <bool OUT_BF16, bool TRANS_OUT>
__global__ __launch_bounds__(256) void gemm_bf16_kernel(
    const __bf16* __restrict__ A, const __bf16* __restrict__ Bsw,
    const float* __restrict__ bias, void* __restrict__ C,
    int M, int Nc, int K) {
  __shared__ __align__(32) __bf16 sA[64 * 32];

  const int tid  = threadIdx.x;
  const int wave = tid >> 5, lane = tid & 31;
  const int r = lane & 15, half = lane >> 4;
  const int m0 = blockIdx.y * 64, n0 = blockIdx.x * 128;
  const int mw = (wave & 1) * 32, nw = (wave >> 1) * 32;
  const int nKs = K >> 5;
  const int cg0 = (n0 + nw) >> 4;      // col-group of this wave's first tile

  // A staging: one 16B chunk per thread per K-step (64x32 bf16 tile).
  const int arow = tid >> 2, aseg = tid & 3;
  const __bf16* ag = A + (size_t)(m0 + arow) * K + aseg * 8;
  __bf16* al = sA + tid * 8;

  f32x8 acc[2][2] = {};

  for (int ks = 0; ks < nKs; ++ks) {
    async_copy16(al, ag);
    ag += 32;
    async_wait();
    __syncthreads();

    bf16x16 bf0 = load_b_frag_packed(Bsw, cg0 + 0, ks, nKs, lane);
    bf16x16 bf1 = load_b_frag_packed(Bsw, cg0 + 1, ks, nKs, lane);
#pragma unroll
    for (int i = 0; i < 2; ++i) {
      bf16x16 af = load_a_frag(sA, mw + i * 16, 32, 0, r, half);
      acc[i][0] = wmma_bf16(af, bf0, acc[i][0]);
      acc[i][1] = wmma_bf16(af, bf1, acc[i][1]);
    }
    __syncthreads();
  }

#pragma unroll
  for (int i = 0; i < 2; ++i)
#pragma unroll
    for (int j = 0; j < 2; ++j) {
      int col = n0 + nw + j * 16 + r;
      float bv = bias ? bias[col] : 0.0f;
#pragma unroll
      for (int v = 0; v < 8; ++v) {
        int row = m0 + mw + i * 16 + half * 8 + v;
        float val = acc[i][j][v] + bv;
        if (TRANS_OUT)
          ((__bf16*)C)[(size_t)col * M + row] = f2bf(val);
        else if (OUT_BF16)
          ((__bf16*)C)[(size_t)row * Nc + col] = f2bf(val);
        else
          ((float*)C)[(size_t)row * Nc + col] = val;
      }
    }
}

// ------------------------- int4 quant-dequant ------------------------------

__global__ __launch_bounds__(256) void kv_amax_kernel(
    const float* __restrict__ kv, float* __restrict__ amax) {
  __shared__ float red[256];
  int c = blockIdx.x;                 // channel 0..R-1
  float m = 0.0f;
  for (int i = threadIdx.x; i < M_; i += 256)
    m = fmaxf(m, fabsf(kv[(size_t)i * R_ + c]));
  red[threadIdx.x] = m;
  __syncthreads();
  for (int s = 128; s > 0; s >>= 1) {
    if (threadIdx.x < s)
      red[threadIdx.x] = fmaxf(red[threadIdx.x], red[threadIdx.x + s]);
    __syncthreads();
  }
  if (threadIdx.x == 0) amax[c] = red[0];
}

__global__ __launch_bounds__(256) void kv_quantdq_kernel(
    const float* __restrict__ kv, const float* __restrict__ amax,
    __bf16* __restrict__ out) {
  int i = blockIdx.x * 256 + threadIdx.x;
  if (i >= M_ * R_) return;
  int c = i & (R_ - 1);
  float a = fmaxf(amax[c], 1e-5f);
  float scale = a * (1.0f / 7.0f);
  float q = fminf(fmaxf(rintf(kv[i] / scale), -8.0f), 7.0f);
  out[i] = f2bf(q * scale);
}

// --------------------------- flash attention -------------------------------
// One block per (b, h, 64-query tile). 256 threads = 8 waves.
// S tiles: wave -> rows (wave&3)*16, cols (wave>>2)*32 + {0,16}
// O tiles: wave -> rows (wave&3)*16, cols (wave>>2)*64 + {0,16,32,48}
// K^T and V staged to LDS in WMMA B-fragment order, pure 16B async copies
// (V comes from the transposed V^T tensor so its fragment runs are
// contiguous in global memory too).

__global__ __launch_bounds__(256) void mla_attention_kernel(
    const __bf16* __restrict__ q, const __bf16* __restrict__ k,
    const __bf16* __restrict__ vt, __bf16* __restrict__ ctx) {
  __shared__ __align__(32) __bf16 sQ[64 * HD_];   // row-major A operand
  __shared__ __align__(32) __bf16 sKf[64 * HD_];  // K^T, frag order (nKs=4, cg=4)
  __shared__ __align__(32) __bf16 sVf[64 * HD_];  // V,   frag order (nKs=2, cg=8)
  __shared__ __align__(32) float  sS[64 * 64];
  __shared__ __align__(32) __bf16 sP[64 * 64];    // row-major A operand
  __shared__ float sM[64], sL[64], sF[64];

  const int tid  = threadIdx.x;
  const int wave = tid >> 5, lane = tid & 31;
  const int r = lane & 15, half = lane >> 4;

  const int ntiles = N_ / 64;                       // 32
  const int bid = blockIdx.x;
  const int qt  = bid % ntiles;
  const int h   = (bid / ntiles) % H_;
  const int b   = bid / (ntiles * H_);
  const int q0  = qt * 64;
  const size_t rowbase = (size_t)b * N_;

  // ---- stage Q (128 rows x 64 dims, row stride 64): 4x 16B chunks per thread ----
#pragma unroll
  for (int i = 0; i < 4; ++i) {
    int cidx = tid + i * 256;                       // 1024 chunks total
    int row = cidx >> 3, seg = cidx & 7;
    async_copy16(sQ + cidx * 8,
                 q + (rowbase + q0 + row) * D_ + h * HD_ + seg * 8);
  }
  if (tid < 64) { sM[tid] = -1e30f; sL[tid] = 0.0f; }

  // ---- precompute per-thread staging addresses for K^T and V ----
  // K^T frag chunk cidx: hh=cidx&1, ln=(cidx>>1)&31, ks=(cidx>>6)&3, cg=cidx>>8
  //   key = cg*16 + (ln&15); kdim = ks*32 + (ln>>4)*16 + hh*8  (16B contiguous)
  const __bf16* kg[4];
  __bf16* kl[4];
  // V frag chunk cidx: hh=cidx&1, ln=(cidx>>1)&31, ks=(cidx>>6)&1, cg=cidx>>7
  //   dim = cg*16 + (ln&15); key = ks*32 + (ln>>4)*16 + hh*8   (16B in V^T)
  const __bf16* vg[4];
  __bf16* vl[4];
#pragma unroll
  for (int i = 0; i < 4; ++i) {
    int cidx = tid + i * 256;
    {
      int hh = cidx & 1, ln = (cidx >> 1) & 31, ks = (cidx >> 6) & 3,
          cg = cidx >> 8;
      int key = cg * 16 + (ln & 15);
      int kd  = ks * 32 + ((ln >> 4) & 1) * 16 + hh * 8;
      kg[i] = k + (rowbase + key) * D_ + h * HD_ + kd;
      kl[i] = sKf + cidx * 8;
    }
    {
      int hh = cidx & 1, ln = (cidx >> 1) & 31, ks = (cidx >> 6) & 1,
          cg = cidx >> 7;
      int dim = cg * 16 + (ln & 15);
      int key = ks * 32 + ((ln >> 4) & 1) * 16 + hh * 8;
      vg[i] = vt + (size_t)(h * HD_ + dim) * M_ + (size_t)b * N_ + key;
      vl[i] = sVf + cidx * 8;
    }
  }

  const int mt    = (wave & 3) * 16;
  const int nbase = (wave >> 2) * 64;
  const int sn    = (wave >> 2) * 32;
  f32x8 o[4] = {};

  async_wait();           // Q tile landed
  __syncthreads();

  const float scale = 0.08838834764831845f;         // 1/sqrt(128)

  for (int kt = 0; kt < ntiles; ++kt) {
    // ---- stage K^T and V tiles: 8x 16B async copies per thread ----
#pragma unroll
    for (int i = 0; i < 4; ++i) {
      async_copy16(kl[i], kg[i]);
      kg[i] += 64 * D_;
      async_copy16(vl[i], vg[i]);
      vg[i] += 64;
    }
    async_wait();
    __syncthreads();

    // S = Q K^T
#pragma unroll
    for (int j = 0; j < 2; ++j) {
      const int cg = (sn >> 4) + j;
      f32x8 s = {};
#pragma unroll
      for (int ks = 0; ks < 4; ++ks) {
        bf16x16 af = load_a_frag(sQ, mt, HD_, ks * 32, r, half);
        bf16x16 bf = load_b_frag_packed(sKf, cg, ks, 4, lane);
        s = wmma_bf16(af, bf, s);
      }
#pragma unroll
      for (int vv = 0; vv < 8; ++vv)
        sS[(mt + half * 8 + vv) * 64 + cg * 16 + r] = s[vv] * scale;
    }
    __syncthreads();

    // online softmax, one thread per query row
    if (tid < 64) {
      float mo = sM[tid];
      float tmax = -1e30f;
      for (int c = 0; c < 64; ++c) tmax = fmaxf(tmax, sS[tid * 64 + c]);
      float mn = fmaxf(mo, tmax);
      float f  = __expf(mo - mn);
      float ls = 0.0f;
      for (int c = 0; c < 64; ++c) {
        float p = __expf(sS[tid * 64 + c] - mn);
        sP[tid * 64 + c] = f2bf(p);
        ls += p;
      }
      sL[tid] = sL[tid] * f + ls;
      sM[tid] = mn;
      sF[tid] = f;
    }
    __syncthreads();

    // O = O*diag(f) + P @ V
#pragma unroll
    for (int j = 0; j < 4; ++j)
#pragma unroll
      for (int vv = 0; vv < 8; ++vv) o[j][vv] *= sF[mt + half * 8 + vv];

#pragma unroll
    for (int ks = 0; ks < 2; ++ks) {
      bf16x16 af = load_a_frag(sP, mt, 64, ks * 32, r, half);
#pragma unroll
      for (int j = 0; j < 4; ++j) {
        const int cg = (nbase >> 4) + j;
        bf16x16 bf = load_b_frag_packed(sVf, cg, ks, 2, lane);
        o[j] = wmma_bf16(af, bf, o[j]);
      }
    }
    __syncthreads();
  }

  // normalize by 1/l and store bf16 context
#pragma unroll
  for (int j = 0; j < 4; ++j) {
    const int no = nbase + j * 16;
#pragma unroll
    for (int vv = 0; vv < 8; ++vv) {
      const int row = mt + half * 8 + vv;
      float inv = 1.0f / sL[row];
      ctx[(rowbase + q0 + row) * D_ + h * HD_ + no + r] = f2bf(o[j][vv] * inv);
    }
  }
}

// ------------------------------- launch ------------------------------------

extern "C" void kernel_launch(void* const* d_in, const int* in_sizes, int n_in,
                              void* d_out, int out_size, void* d_ws,
                              size_t ws_size, hipStream_t stream) {
  (void)in_sizes; (void)n_in; (void)out_size; (void)ws_size;

  const float* x   = (const float*)d_in[0];
  const float* Wq  = (const float*)d_in[1];
  const float* bq  = (const float*)d_in[2];
  const float* Wkv = (const float*)d_in[3];
  const float* bkv = (const float*)d_in[4];
  const float* Wku = (const float*)d_in[5];
  const float* bku = (const float*)d_in[6];
  const float* Wvu = (const float*)d_in[7];
  const float* bvu = (const float*)d_in[8];
  const float* Wo  = (const float*)d_in[9];
  const float* bo  = (const float*)d_in[10];

  // workspace bump allocator (256B aligned)
  char* w = (char*)d_ws;
  auto alloc = [&](size_t bytes) -> void* {
    void* p = (void*)w;
    w += (bytes + 255) & ~(size_t)255;
    return p;
  };
  __bf16* xb    = (__bf16*)alloc((size_t)M_ * D_ * 2);
  __bf16* Wqb   = (__bf16*)alloc((size_t)D_ * D_ * 2);   // fragment order
  __bf16* Wkvb  = (__bf16*)alloc((size_t)D_ * R_ * 2);   // fragment order
  __bf16* Wkub  = (__bf16*)alloc((size_t)R_ * D_ * 2);   // fragment order
  __bf16* Wvub  = (__bf16*)alloc((size_t)R_ * D_ * 2);   // fragment order
  __bf16* Wob   = (__bf16*)alloc((size_t)D_ * D_ * 2);   // fragment order
  __bf16* qb    = (__bf16*)alloc((size_t)M_ * D_ * 2);
  float*  kvf   = (float*) alloc((size_t)M_ * R_ * 4);
  float*  amax  = (float*) alloc((size_t)R_ * 4);
  __bf16* kvdqb = (__bf16*)alloc((size_t)M_ * R_ * 2);
  __bf16* kb    = (__bf16*)alloc((size_t)M_ * D_ * 2);
  __bf16* vbt   = (__bf16*)alloc((size_t)M_ * D_ * 2);   // V^T (dim-major)
  __bf16* ctxb  = (__bf16*)alloc((size_t)M_ * D_ * 2);

  // x stays row-major (A operand)
  f32_to_bf16_kernel<<<(M_ * D_ + 255) / 256, 256, 0, stream>>>(x, xb, M_ * D_);
  // weights go to fragment order (B operands)
  auto swz = [&](const float* src, __bf16* dst, int K, int Nc) {
    f32_to_bf16_swzB_kernel<<<(K * Nc + 255) / 256, 256, 0, stream>>>(src, dst,
                                                                      K, Nc);
  };
  swz(Wq,  Wqb,  D_, D_);
  swz(Wkv, Wkvb, D_, R_);
  swz(Wku, Wkub, R_, D_);
  swz(Wvu, Wvub, R_, D_);
  swz(Wo,  Wob,  D_, D_);

  // q = x @ Wq + bq  -> bf16
  gemm_bf16_kernel<true, false><<<dim3(D_ / 128, M_ / 64), 256, 0, stream>>>(
      xb, Wqb, bq, qb, M_, D_, D_);
  // kv = x @ Wkv + bkv -> f32
  gemm_bf16_kernel<false, false><<<dim3(R_ / 128, M_ / 64), 256, 0, stream>>>(
      xb, Wkvb, bkv, kvf, M_, R_, D_);

  // int4 quant-dequant (per-channel amax over B*N)
  kv_amax_kernel<<<R_, 256, 0, stream>>>(kvf, amax);
  kv_quantdq_kernel<<<(M_ * R_ + 255) / 256, 256, 0, stream>>>(kvf, amax, kvdqb);

  // k = kv_dq @ Wku + bku -> bf16 ; v = kv_dq @ Wvu + bvu -> bf16 transposed
  gemm_bf16_kernel<true, false><<<dim3(D_ / 128, M_ / 64), 256, 0, stream>>>(
      kvdqb, Wkub, bku, kb, M_, D_, R_);
  gemm_bf16_kernel<true, true><<<dim3(D_ / 128, M_ / 64), 256, 0, stream>>>(
      kvdqb, Wvub, bvu, vbt, M_, D_, R_);

  // attention -> ctx (bf16)
  mla_attention_kernel<<<B_ * H_ * (N_ / 64), 256, 0, stream>>>(qb, kb, vbt,
                                                                ctxb);

  // out = ctx @ Wo + bo -> f32 d_out
  gemm_bf16_kernel<false, false><<<dim3(D_ / 128, M_ / 64), 256, 0, stream>>>(
      ctxb, Wob, bo, (float*)d_out, M_, D_, D_);
}
